// SelfAttention_v2_10428180595253
// MI455X (gfx1250) — matrix operand order repaired and checked by reading, not compiled, not run on hardware
//
#include <hip/hip_runtime.h>
#include <stdint.h>

#define SEQ  8192
#define DIN  768
#define DOUT 64

typedef __attribute__((ext_vector_type(16))) __bf16 v16bf;
typedef __attribute__((ext_vector_type(2)))  __bf16 v2bf;
typedef __attribute__((ext_vector_type(8)))  float  v8f;

union Frag {
  v16bf v;
  uint4 q[2];
  uint32_t u[8];
};

// Scalar RNE f32->bf16 (epilogue stores only)
__device__ __forceinline__ uint32_t f2bf(float f) {
  uint32_t u = __float_as_uint(f);
  u += 0x7fffu + ((u >> 16) & 1u);
  return u >> 16;
}

// Packed f32x2 -> bf16x2 using the HW convert (v_cvt_pk_bf16_f32)
__device__ __forceinline__ uint32_t pk2bf(float lo, float hi) {
#if __has_builtin(__builtin_amdgcn_cvt_pk_bf16_f32)
  v2bf v = __builtin_amdgcn_cvt_pk_bf16_f32(lo, hi);
#else
  v2bf v = {(__bf16)lo, (__bf16)hi};
#endif
  uint32_t r;
  __builtin_memcpy(&r, &v, 4);
  return r;
}

__device__ __forceinline__ v16bf load_frag(const uint16_t* p0, const uint16_t* p1) {
  Frag f;
  f.q[0] = *reinterpret_cast<const uint4*>(p0);
  f.q[1] = *reinterpret_cast<const uint4*>(p1);
  return f.v;
}
__device__ __forceinline__ v8f zero8() {
  v8f z = {0.f, 0.f, 0.f, 0.f, 0.f, 0.f, 0.f, 0.f};
  return z;
}

// ---- async global->LDS staging (ASYNCcnt) with sync fallback -------------
#if __has_builtin(__builtin_amdgcn_global_load_async_to_lds_b128)
#define ASYNC_STAGE 1
// Signature (from clang diagnostic): param0 = int4 in global AS, then LDS
// dst, imm offset, imm cpol.
typedef __attribute__((__vector_size__(16))) int vsi4;
typedef vsi4 __attribute__((address_space(1)))* gv4p;
typedef vsi4 __attribute__((address_space(3)))* lv4p;
#else
#define ASYNC_STAGE 0
#endif

__device__ __forceinline__ void wait_async0() {
#if __has_builtin(__builtin_amdgcn_s_wait_asynccnt)
  __builtin_amdgcn_s_wait_asynccnt(0);
#else
  asm volatile("s_wait_asynccnt 0x0" ::: "memory");
#endif
}

// copy 64 contiguous bytes global->LDS (per thread)
__device__ __forceinline__ void stage64(uint16_t* l, const uint16_t* g) {
#if ASYNC_STAGE
  __builtin_amdgcn_global_load_async_to_lds_b128((gv4p)g, (lv4p)l, 0, 0);
  __builtin_amdgcn_global_load_async_to_lds_b128((gv4p)g, (lv4p)l, 16, 0);
  __builtin_amdgcn_global_load_async_to_lds_b128((gv4p)g, (lv4p)l, 32, 0);
  __builtin_amdgcn_global_load_async_to_lds_b128((gv4p)g, (lv4p)l, 48, 0);
#else
#pragma unroll
  for (int i = 0; i < 4; ++i)
    *reinterpret_cast<uint4*>(l + i * 8) =
        *reinterpret_cast<const uint4*>(g + i * 8);
#endif
}

// ---------------------------------------------------------------------------
// Kernel 1: W [768x64] f32 -> Wt [64x768] bf16.
// Wq is pre-scaled by (1/sqrt(64)) * log2(e) so attention runs in base-2.
// ---------------------------------------------------------------------------
__global__ void prep_w(const float* __restrict__ Wq, const float* __restrict__ Wk,
                       const float* __restrict__ Wv,
                       uint16_t* __restrict__ WtQ, uint16_t* __restrict__ WtK,
                       uint16_t* __restrict__ WtV) {
  int idx = blockIdx.x * blockDim.x + threadIdx.x;
  if (idx >= DIN * DOUT) return;
  int n = idx / DIN, k = idx % DIN;
  const float qs = 0.125f * 1.4426950408889634f;
  WtQ[n * DIN + k] = (uint16_t)f2bf(Wq[k * DOUT + n] * qs);
  WtK[n * DIN + k] = (uint16_t)f2bf(Wk[k * DOUT + n]);
  WtV[n * DIN + k] = (uint16_t)f2bf(Wv[k * DOUT + n]);
}

// ---------------------------------------------------------------------------
// Kernel 2: Q,K row-major bf16 [SEQ][64]; Vt = V^T bf16 [64][SEQ].
// One 16-row strip per wave, 12 WMMA bf16 16x16x32 per K-step.
// ---------------------------------------------------------------------------
__global__ void __launch_bounds__(256) proj_qkv(
    const float* __restrict__ x,
    const uint16_t* __restrict__ WtQ, const uint16_t* __restrict__ WtK,
    const uint16_t* __restrict__ WtV,
    uint16_t* __restrict__ Qb, uint16_t* __restrict__ Kb,
    uint16_t* __restrict__ Vt) {
  const int lane = threadIdx.x & 31;
  const int wid  = threadIdx.x >> 5;
  const int rb   = blockIdx.x * 128 + wid * 16;
  const int l15  = lane & 15;
  const int hi   = lane >> 4;
  const uint16_t* wmat[3] = {WtQ, WtK, WtV};

  v8f acc[3][4];
#pragma unroll
  for (int m = 0; m < 3; ++m)
#pragma unroll
    for (int t = 0; t < 4; ++t) acc[m][t] = zero8();

  const float* xr = x + (size_t)(rb + l15) * DIN;
  for (int k0 = 0; k0 < DIN; k0 += 32) {
    const float* p = xr + k0 + hi * 8;
    float4 f0 = *reinterpret_cast<const float4*>(p);
    float4 f1 = *reinterpret_cast<const float4*>(p + 4);
    float4 f2 = *reinterpret_cast<const float4*>(p + 16);
    float4 f3 = *reinterpret_cast<const float4*>(p + 20);
    Frag a;
    a.u[0] = pk2bf(f0.x, f0.y); a.u[1] = pk2bf(f0.z, f0.w);
    a.u[2] = pk2bf(f1.x, f1.y); a.u[3] = pk2bf(f1.z, f1.w);
    a.u[4] = pk2bf(f2.x, f2.y); a.u[5] = pk2bf(f2.z, f2.w);
    a.u[6] = pk2bf(f3.x, f3.y); a.u[7] = pk2bf(f3.z, f3.w);
#pragma unroll
    for (int m = 0; m < 3; ++m) {
#pragma unroll
      for (int t = 0; t < 4; ++t) {
        const uint16_t* w = wmat[m] + (size_t)(t * 16 + l15) * DIN + k0 + hi * 8;
        v16bf b = load_frag(w, w + 16);
        acc[m][t] = __builtin_amdgcn_wmma_f32_16x16x32_bf16(
            false, a.v, false, b, (short)0, acc[m][t], false, false);
      }
    }
  }
#pragma unroll
  for (int t = 0; t < 4; ++t) {
#pragma unroll
    for (int j = 0; j < 8; ++j) {
      int row = rb + j + hi * 8;
      int col = t * 16 + l15;
      Qb[(size_t)row * DOUT + col] = (uint16_t)f2bf(acc[0][t][j]);
      Kb[(size_t)row * DOUT + col] = (uint16_t)f2bf(acc[1][t][j]);
      Vt[(size_t)col * SEQ + row]  = (uint16_t)f2bf(acc[2][t][j]);
    }
  }
}

// ---------------------------------------------------------------------------
// Kernel 3: fused flash attention with double-buffered async LDS staging.
// Per 128-key chunk the block DMAs K (128x64) and V^T (64x128) into the
// idle LDS buffer (global_load_async_to_lds, ASYNCcnt) while computing the
// current chunk from the other buffer; one barrier per chunk.  Scores are
// computed transposed (S^T = K @ Q^T) so online-softmax stats are lane-
// local and S^T(D-layout) -> P(A-layout) needs no cross-lane traffic.
// ---------------------------------------------------------------------------
#define KT   128          // keys staged per chunk
#define KPAD 72           // kbuf row stride: bank start L*36%64, all distinct
#define VPAD 136          // vbuf row stride: bank start L*4 %64, all distinct

__global__ void __launch_bounds__(256) flash_attn(
    const uint16_t* __restrict__ Qb, const uint16_t* __restrict__ Kb,
    const uint16_t* __restrict__ Vt, float* __restrict__ out) {
  __shared__ uint16_t kbuf[2][KT * KPAD];     // 2 x 18432 B
  __shared__ uint16_t vbuf[2][DOUT * VPAD];   // 2 x 17408 B

  const int tid  = threadIdx.x;
  const int lane = tid & 31;
  const int wid  = tid >> 5;
  const int qb   = blockIdx.x * 128 + wid * 16;
  const int l15  = lane & 15;
  const int hi   = lane >> 4;
  const int srcbase = hi * 8;

  // Q^T B-fragments, held live across the whole key loop
  const uint16_t* qp = Qb + (size_t)(qb + l15) * DOUT + hi * 8;
  v16bf bq0 = load_frag(qp, qp + 16);         // kd 0..31
  v16bf bq1 = load_frag(qp + 32, qp + 48);    // kd 32..63

  // staging assignment: 256 threads x 64B for K, 64B for V
  const int krow = tid >> 1, khalf = (tid & 1) * 32;    // K: 128 rows x 64
  const int vrow = tid >> 2, vq    = (tid & 3) * 32;    // V: 64 rows x 128

  auto stage = [&](int buf, int kb2) {
    stage64(&kbuf[buf][krow * KPAD + khalf],
            Kb + (size_t)(kb2 + krow) * DOUT + khalf);
    stage64(&vbuf[buf][vrow * VPAD + vq],
            Vt + (size_t)vrow * SEQ + kb2 + vq);
  };

  v8f o[4];
#pragma unroll
  for (int t = 0; t < 4; ++t) o[t] = zero8();
  float mrun = -1e30f;
  float lrun = 0.0f;

  stage(0, 0);
  wait_async0();
  __syncthreads();

  for (int kb = 0; kb < SEQ; kb += KT) {
    const int cur = (kb / KT) & 1;
    if (kb + KT < SEQ) stage(cur ^ 1, kb + KT);   // prefetch next chunk

    const uint16_t* kcur = kbuf[cur];
    const uint16_t* vcur = vbuf[cur];
#pragma unroll
    for (int sub = 0; sub < KT / 32; ++sub) {
      const int kl = sub * 32;
      // ---- S^T tiles: 2 x (16 keys x 16 queries), kd=64 -> 2 WMMAs each
      v8f st[2];
#pragma unroll
      for (int t = 0; t < 2; ++t) {
        const uint16_t* kp = kcur + (kl + t * 16 + l15) * KPAD + hi * 8;
        v16bf a0 = load_frag(kp,      kp + 16);
        v16bf a1 = load_frag(kp + 32, kp + 48);
        v8f c = zero8();
        c = __builtin_amdgcn_wmma_f32_16x16x32_bf16(false, a0, false, bq0,
                                                    (short)0, c, false, false);
        c = __builtin_amdgcn_wmma_f32_16x16x32_bf16(false, a1, false, bq1,
                                                    (short)0, c, false, false);
        st[t] = c;
      }
      // ---- online softmax (base 2), per query = per lane ----
      float mx = mrun;
#pragma unroll
      for (int t = 0; t < 2; ++t)
#pragma unroll
        for (int j = 0; j < 8; ++j) mx = fmaxf(mx, st[t][j]);
      mx = fmaxf(mx, __shfl_xor(mx, 16));
      float alpha = exp2f(mrun - mx);
      mrun = mx;
      float s = 0.0f;
#pragma unroll
      for (int t = 0; t < 2; ++t)
#pragma unroll
        for (int j = 0; j < 8; ++j) {
          float pe = exp2f(st[t][j] - mx);
          st[t][j] = pe;
          s += pe;
        }
      s += __shfl_xor(s, 16);
      lrun = lrun * alpha + s;

      // ---- pack P: S^T D-layout -> P A-layout, fully lane-local ----
      Frag pf;
#pragma unroll
      for (int v = 0; v < 4; ++v) {
        pf.u[v]     = pk2bf(st[0][2 * v], st[0][2 * v + 1]);
        pf.u[4 + v] = pk2bf(st[1][2 * v], st[1][2 * v + 1]);
      }

      // ---- rescale O by alpha (broadcast lane -> accumulator row) ----
#pragma unroll
      for (int j = 0; j < 8; ++j) {
        float aj = __shfl(alpha, j + srcbase);
#pragma unroll
        for (int t = 0; t < 4; ++t) o[t][j] *= aj;
      }

      // ---- O += P @ V ----
#pragma unroll
      for (int t = 0; t < 4; ++t) {
        const uint16_t* vp = vcur + (t * 16 + l15) * VPAD + kl + hi * 8;
        v16bf bv = load_frag(vp, vp + 16);
        o[t] = __builtin_amdgcn_wmma_f32_16x16x32_bf16(false, pf.v, false, bv,
                                                       (short)0, o[t], false, false);
      }
    }
    wait_async0();     // this wave's DMA into buf[cur^1] has landed
    __syncthreads();   // everyone done reading buf[cur] & staging buf[cur^1]
  }

  // ---- normalize and write fp32 output ----
  float inv = 1.0f / lrun;
#pragma unroll
  for (int j = 0; j < 8; ++j) {
    float rj = __shfl(inv, j + srcbase);
    int row = qb + j + srcbase;
#pragma unroll
    for (int t = 0; t < 4; ++t)
      out[(size_t)row * DOUT + t * 16 + l15] = o[t][j] * rj;
  }
}

// ---------------------------------------------------------------------------
extern "C" void kernel_launch(void* const* d_in, const int* in_sizes, int n_in,
                              void* d_out, int out_size, void* d_ws, size_t ws_size,
                              hipStream_t stream) {
  (void)in_sizes; (void)n_in; (void)out_size; (void)ws_size;
  const float* x  = (const float*)d_in[0];
  const float* Wq = (const float*)d_in[1];
  const float* Wk = (const float*)d_in[2];
  const float* Wv = (const float*)d_in[3];
  float* out = (float*)d_out;

  char* ws = (char*)d_ws;
  size_t off = 0;
  auto alloc = [&](size_t bytes) -> void* {
    void* p = ws + off;
    off = (off + bytes + 255) & ~(size_t)255;
    return p;
  };
  uint16_t* Qb  = (uint16_t*)alloc((size_t)SEQ * DOUT * 2);
  uint16_t* Kb  = (uint16_t*)alloc((size_t)SEQ * DOUT * 2);
  uint16_t* Vt  = (uint16_t*)alloc((size_t)SEQ * DOUT * 2);
  uint16_t* WtQ = (uint16_t*)alloc((size_t)DIN * DOUT * 2);
  uint16_t* WtK = (uint16_t*)alloc((size_t)DIN * DOUT * 2);
  uint16_t* WtV = (uint16_t*)alloc((size_t)DIN * DOUT * 2);

  prep_w<<<(DIN * DOUT + 255) / 256, 256, 0, stream>>>(Wq, Wk, Wv, WtQ, WtK, WtV);
  proj_qkv<<<SEQ / 128, 256, 0, stream>>>(x, WtQ, WtK, WtV, Qb, Kb, Vt);
  flash_attn<<<SEQ / 128, 256, 0, stream>>>(Qb, Kb, Vt, out);
}
